// RNA_feature_extraction_57870389347015
// MI455X (gfx1250) — matrix operand order repaired and checked by reading, not compiled, not run on hardware
//
#include <hip/hip_runtime.h>
#include <hip/hip_bf16.h>
#include <stdint.h>

#define HS    256
#define CIN   256
#define COUT  128
#define BSZ   256
#define LSEQ  400
#define SPAD  512
#define NTOT  (BSZ * LSEQ)     // 102400
#define KEMB  640
#define TAPS  33

typedef __bf16 bf16;
typedef bf16  v16bf  __attribute__((ext_vector_type(16)));
typedef bf16  bf16x8 __attribute__((ext_vector_type(8)));
typedef float v8f    __attribute__((ext_vector_type(8)));
typedef unsigned int u32x4 __attribute__((ext_vector_type(4)));
typedef int   i32x8  __attribute__((ext_vector_type(8)));
typedef int   i32x4  __attribute__((ext_vector_type(4)));

union FragAB { v16bf v; bf16x8 h[2]; };

__device__ __forceinline__ bf16 f2bf(float f) {
  uint32_t u = __builtin_bit_cast(uint32_t, f);
  u += 0x7FFFu + ((u >> 16) & 1u);              // round-to-nearest-even
  uint16_t s = (uint16_t)(u >> 16);
  return __builtin_bit_cast(bf16, s);
}

__device__ __forceinline__ v8f wmma_bf16(const FragAB& a, const FragAB& b, v8f c) {
  return __builtin_amdgcn_wmma_f32_16x16x32_bf16(false, a.v, false, b.v, (short)0, c,
                                                 false, false);
}

// Load one lane's A or B fragment: two 16-byte chunks, second at +offBytes.
__device__ __forceinline__ FragAB load_frag(const bf16* p, int offBytes) {
  FragAB f;
  f.h[0] = *reinterpret_cast<const bf16x8*>(p);
  f.h[1] = *reinterpret_cast<const bf16x8*>(reinterpret_cast<const char*>(p) + offBytes);
  return f;
}

// ---------------------------------------------------------------------------
// Prep: convert/transpose weights to bf16 layouts, fold /3 into conv weights.
// ---------------------------------------------------------------------------
__global__ void prep_kernel(const float* __restrict__ k7,  const float* __restrict__ k11,
                            const float* __restrict__ k15, const float* __restrict__ cb7,
                            const float* __restrict__ cb11,const float* __restrict__ cb15,
                            const float* __restrict__ W_le,const float* __restrict__ Wl1,
                            const float* __restrict__ Wl2,
                            bf16* __restrict__ wt, float* __restrict__ bsum,
                            bf16* __restrict__ wle_t, bf16* __restrict__ wl1t,
                            bf16* __restrict__ wl2t) {
  const int gt = blockIdx.x * blockDim.x + threadIdx.x;
  const int gs = gridDim.x * blockDim.x;
  const float inv3 = 1.0f / 3.0f;
  for (int i = gt; i < TAPS * COUT * CIN; i += gs) {
    int t = i / (COUT * CIN);
    int rem = i - t * (COUT * CIN);
    int o = rem >> 8, c = rem & 255;
    float v;
    if (t < 7)       v = k7 [(o * CIN + c) * 7  + t];
    else if (t < 18) v = k11[(o * CIN + c) * 11 + (t - 7)];
    else             v = k15[(o * CIN + c) * 15 + (t - 18)];
    wt[i] = f2bf(v * inv3);
  }
  for (int i = gt; i < COUT; i += gs) bsum[i] = (cb7[i] + cb11[i] + cb15[i]) * inv3;
  for (int i = gt; i < HS * KEMB; i += gs) {
    int n = i / KEMB, k = i - n * KEMB;
    wle_t[i] = f2bf(W_le[k * HS + n]);
  }
  for (int i = gt; i < 512 * COUT; i += gs) {
    int n = i >> 7, k = i & 127;
    wl1t[i] = f2bf(Wl1[k * 512 + n]);
  }
  for (int i = gt; i < HS * 512; i += gs) {
    int n = i >> 9, k = i & 511;
    wl2t[i] = f2bf(Wl2[k * HS + n]);
  }
}

// Zero sequence padding region h[:, 400:512, :]  (h is [B][SPAD][HS])
__global__ void zeropad_kernel(bf16* __restrict__ h) {
  const int PADELEMS = (SPAD - LSEQ) * HS;  // 28672
  int i = blockIdx.x * blockDim.x + threadIdx.x;
  if (i < BSZ * PADELEMS) {
    int b = i / PADELEMS, rem = i - b * PADELEMS;
    h[((size_t)b * SPAD + LSEQ) * HS + rem] = __builtin_bit_cast(bf16, (uint16_t)0);
  }
}

// out[b][c] = (400/512)*bl2[c]  (bias contribution of the masked mean)
__global__ void initout_kernel(float* __restrict__ out, const float* __restrict__ bl2) {
  int i = blockIdx.x * blockDim.x + threadIdx.x;
  if (i < BSZ * HS) out[i] = (400.0f / 512.0f) * bl2[i & 255];
}

// ---------------------------------------------------------------------------
// K1: h[b][s][c] = 0.5*(relu(emb@W_le + b_le) + emb_table[x_tokens])   (bf16)
// One workgroup = 16 rows x 256 cols; 8 waves x 2 N-tiles; K=640 -> 20 chunks.
// ---------------------------------------------------------------------------
__global__ __launch_bounds__(256) void k1_embed(const float* __restrict__ emb,
                                                const int*   __restrict__ x_tokens,
                                                const float* __restrict__ emb_table,
                                                const bf16*  __restrict__ wle_t,
                                                const float* __restrict__ b_le,
                                                bf16* __restrict__ h) {
  __shared__ __align__(16) bf16 As[16 * KEMB];  // 20 KB, row-major [16][640]
  const int tid  = threadIdx.x;
  const int row0 = blockIdx.x * 16;
  for (int i = tid; i < 16 * KEMB; i += 256) {
    int r = i / KEMB, k = i - r * KEMB;
    As[i] = f2bf(emb[(size_t)(row0 + r) * KEMB + k]);
  }
  __syncthreads();

  const int wave = tid >> 5, lane = tid & 31;
  const int hi = lane >> 4, nl = lane & 15;
  const int n0 = wave * 32;
  v8f acc0 = {}, acc1 = {};
  const bf16* aBase = &As[nl * KEMB + hi * 16];
  const bf16* b0 = wle_t + (size_t)(n0 + nl) * KEMB + hi * 16;
  const bf16* b1 = wle_t + (size_t)(n0 + 16 + nl) * KEMB + hi * 16;
#pragma unroll 4
  for (int kc = 0; kc < KEMB / 32; ++kc) {
    FragAB a = load_frag(aBase + kc * 32, 16);
    FragAB bb0 = load_frag(b0 + kc * 32, 16);
    acc0 = wmma_bf16(a, bb0, acc0);
    FragAB bb1 = load_frag(b1 + kc * 32, 16);
    acc1 = wmma_bf16(a, bb1, acc1);
  }
#pragma unroll
  for (int j = 0; j < 2; ++j) {
    v8f acc = j ? acc1 : acc0;
    int c = n0 + j * 16 + nl;
    float bias = b_le[c];
#pragma unroll
    for (int r = 0; r < 8; ++r) {
      int node = row0 + r + hi * 8;
      float v = acc[r] + bias;
      v = v > 0.0f ? v : 0.0f;
      float xr = emb_table[x_tokens[node] * HS + c];
      int b = node / LSEQ;
      int s = node - b * LSEQ;
      h[((size_t)b * SPAD + s) * HS + c] = f2bf(0.5f * (v + xr));
    }
  }
}

// ---------------------------------------------------------------------------
// K2: fused 3-kernel conv1d as 33 shifted GEMMs.
// Activation slice is DMA'd into LDS by the Tensor Data Mover (one D# per WG):
// 2D tile, tile_dim0 = 256 ch (512 B rows), tile_dim1 = 94 positions.
// ---------------------------------------------------------------------------
__global__ __launch_bounds__(256) void k2_conv(const bf16* __restrict__ h,
                                               const bf16* __restrict__ wt,
                                               const float* __restrict__ bsum,
                                               bf16* __restrict__ yc) {
  __shared__ __align__(16) bf16 xs[96 * 256];   // 48 KB: [pos][chan]
  const int tid = threadIdx.x;
  const int s0 = blockIdx.x * 80;               // 0,80,...,320
  const int b  = blockIdx.y;
  const bf16* hb = h + (size_t)b * SPAD * HS;
  const int wave = tid >> 5, lane = tid & 31;

  const int row0  = (s0 == 0) ? 7 : 0;          // first valid LDS row
  const int rows  = 94 - row0;                  // rows to DMA
  const int p0    = (s0 == 0) ? 0 : (s0 - 7);   // first source position

  if (s0 == 0) {                                 // left halo: zero rows 0..6
    for (int i = tid; i < 7 * 256; i += 256)
      xs[i] = __builtin_bit_cast(bf16, (uint16_t)0);
  }
  if (wave == 0) {
    // ---- Tensor DMA descriptor (D#), groups per CDNA5 ISA ch.8 ----
    uint64_t gaddr = (uint64_t)(uintptr_t)(hb + (size_t)p0 * HS);
    uint32_t laddr = (uint32_t)(uintptr_t)(&xs[row0 * 256]);
    u32x4 g0;
    g0[0] = 1u;                                          // count=1, user mode
    g0[1] = laddr;                                       // lds_addr
    g0[2] = (uint32_t)gaddr;                             // global_addr[31:0]
    g0[3] = (uint32_t)((gaddr >> 32) & 0x01FFFFFFu)      // global_addr[56:32]
          | (2u << 30);                                  // type=2 (image)
    i32x8 g1;
    g1[0] = (int)(1u << 16);                             // data_size=2 bytes
    g1[1] = (int)(256u << 16);                           // tensor_dim0 = 256
    g1[2] = (int)((uint32_t)SPAD << 16);                 // tensor_dim1 = 512
    g1[3] = (int)(256u << 16);                           // tile_dim0 = 256
    g1[4] = rows;                                        // tile_dim1, tile_dim2=0
    g1[5] = 256;                                         // tensor_dim0_stride
    g1[6] = 0;
    g1[7] = 0;
    i32x4 gz4 = {0, 0, 0, 0};                            // <=2D: groups 2,3 unused
    i32x8 gz8 = {0, 0, 0, 0, 0, 0, 0, 0};
    __builtin_amdgcn_tensor_load_to_lds(g0, g1, gz4, gz4, gz8, 0);
  }
  __builtin_amdgcn_s_wait_tensorcnt(0);
  __syncthreads();

  const int hi = lane >> 4, nl = lane & 15;
  const int ob = wave * 16;                     // out-channel tile base
  v8f acc[5] = {};
  const bf16* wlane = wt + (size_t)(ob + nl) * CIN + hi * 16;
#pragma unroll 1
  for (int t = 0; t < TAPS; ++t) {
    int d = (t < 7) ? (t - 3) : (t < 18) ? (t - 12) : (t - 25);
    const bf16* wtap = wlane + (size_t)t * COUT * CIN;
    if (t + 1 < TAPS) __builtin_prefetch(wtap + COUT * CIN, 0, 1);
    const int rbase = d + 7 + nl;               // LDS row for this lane's N
#pragma unroll 2
    for (int kc = 0; kc < CIN / 32; ++kc) {
      FragAB a = load_frag(wtap + kc * 32, 16);
      const bf16* xb = &xs[rbase * 256 + kc * 32 + hi * 16];
#pragma unroll
      for (int st = 0; st < 5; ++st) {
        FragAB bb = load_frag(xb + st * 16 * 256, 16);
        acc[st] = wmma_bf16(a, bb, acc[st]);
      }
    }
  }
  float bs[8];
#pragma unroll
  for (int r = 0; r < 8; ++r) bs[r] = bsum[ob + r + hi * 8];
  bf16* ycb = yc + (size_t)b * LSEQ * COUT;
#pragma unroll
  for (int st = 0; st < 5; ++st) {
    int s = s0 + st * 16 + nl;
#pragma unroll
    for (int r = 0; r < 8; ++r) {
      int o = ob + r + hi * 8;
      ycb[(size_t)s * COUT + o] = f2bf(acc[st][r] + bs[r]);
    }
  }
}

// ---------------------------------------------------------------------------
// K3: z = relu(yc@Wl1 + bl1); partial = z@Wl2; out[b] += sum_rows(partial)/512.
// ---------------------------------------------------------------------------
__global__ __launch_bounds__(256) void k3_mlp(const bf16* __restrict__ yc,
                                              const bf16* __restrict__ wl1t,
                                              const float* __restrict__ bl1,
                                              const bf16* __restrict__ wl2t,
                                              float* __restrict__ out) {
  __shared__ __align__(16) bf16 z1[16 * 512];   // 16 KB
  const int tid = threadIdx.x;
  const int row0 = blockIdx.x * 16;
  const int b = row0 / LSEQ;
  const int wave = tid >> 5, lane = tid & 31;
  const int hi = lane >> 4, nl = lane & 15;

  { // stage 1: M=16, K=128, N=512; 8 waves x 4 N-tiles
    const bf16* aBase = yc + (size_t)(row0 + nl) * COUT + hi * 16;
    const int n0 = wave * 64;
    v8f acc[4] = {};
#pragma unroll
    for (int kc = 0; kc < COUT / 32; ++kc) {
      FragAB a = load_frag(aBase + kc * 32, 16);
#pragma unroll
      for (int j = 0; j < 4; ++j) {
        const bf16* bp = wl1t + (size_t)(n0 + j * 16 + nl) * COUT + kc * 32 + hi * 16;
        FragAB bb = load_frag(bp, 16);
        acc[j] = wmma_bf16(a, bb, acc[j]);
      }
    }
#pragma unroll
    for (int j = 0; j < 4; ++j) {
      int n = n0 + j * 16 + nl;
      float bias = bl1[n];
#pragma unroll
      for (int r = 0; r < 8; ++r) {
        float v = acc[j][r] + bias;
        v = v > 0.0f ? v : 0.0f;
        z1[(r + hi * 8) * 512 + n] = f2bf(v);
      }
    }
  }
  __syncthreads();
  { // stage 2: M=16, K=512, N=256; 8 waves x 2 N-tiles
    const bf16* aB = &z1[nl * 512 + hi * 16];
    const int n0 = wave * 32;
    v8f acc[2] = {};
#pragma unroll 4
    for (int kc = 0; kc < 512 / 32; ++kc) {
      FragAB a = load_frag(aB + kc * 32, 16);
#pragma unroll
      for (int j = 0; j < 2; ++j) {
        const bf16* bp = wl2t + (size_t)(n0 + j * 16 + nl) * 512 + kc * 32 + hi * 16;
        FragAB bb = load_frag(bp, 16);
        acc[j] = wmma_bf16(a, bb, acc[j]);
      }
    }
    const float inv = 1.0f / (float)SPAD;
#pragma unroll
    for (int j = 0; j < 2; ++j) {
      int c = n0 + j * 16 + nl;
      float s = 0.0f;
#pragma unroll
      for (int r = 0; r < 8; ++r) s += acc[j][r];
      atomicAdd(&out[b * HS + c], s * inv);
    }
  }
}

// ---------------------------------------------------------------------------
extern "C" void kernel_launch(void* const* d_in, const int* in_sizes, int n_in,
                              void* d_out, int out_size, void* d_ws, size_t ws_size,
                              hipStream_t stream) {
  (void)in_sizes; (void)n_in; (void)out_size; (void)ws_size;
  const float* emb       = (const float*)d_in[0];
  const int*   x_tokens  = (const int*)  d_in[1];
  const float* emb_table = (const float*)d_in[7];
  const float* W_le      = (const float*)d_in[22];
  const float* b_le      = (const float*)d_in[23];
  const float* k7        = (const float*)d_in[24];
  const float* cb7       = (const float*)d_in[25];
  const float* k11       = (const float*)d_in[26];
  const float* cb11      = (const float*)d_in[27];
  const float* k15       = (const float*)d_in[28];
  const float* cb15      = (const float*)d_in[29];
  const float* Wl1       = (const float*)d_in[30];
  const float* bl1       = (const float*)d_in[31];
  const float* Wl2       = (const float*)d_in[32];
  const float* bl2       = (const float*)d_in[33];
  float* out = (float*)d_out;

  char* ws = (char*)d_ws;
  size_t off = 0;
  auto take = [&](size_t bytes) { char* p = ws + off; off = (off + bytes + 255) & ~(size_t)255; return p; };
  bf16*  h     = (bf16*) take((size_t)BSZ * SPAD * HS * 2);     // 64 MB  [B][S][C]
  bf16*  yc    = (bf16*) take((size_t)BSZ * LSEQ * COUT * 2);   // 25 MB
  bf16*  wt    = (bf16*) take((size_t)TAPS * COUT * CIN * 2);
  bf16*  wle_t = (bf16*) take((size_t)HS * KEMB * 2);
  bf16*  wl1t  = (bf16*) take((size_t)512 * COUT * 2);
  bf16*  wl2t  = (bf16*) take((size_t)HS * 512 * 2);
  float* bsum  = (float*)take((size_t)COUT * 4);

  prep_kernel<<<512, 256, 0, stream>>>(k7, k11, k15, cb7, cb11, cb15,
                                       W_le, Wl1, Wl2, wt, bsum, wle_t, wl1t, wl2t);
  zeropad_kernel<<<(BSZ * (SPAD - LSEQ) * HS + 255) / 256, 256, 0, stream>>>(h);
  initout_kernel<<<(BSZ * HS + 255) / 256, 256, 0, stream>>>(out, bl2);
  k1_embed<<<NTOT / 16, 256, 0, stream>>>(emb, x_tokens, emb_table, wle_t, b_le, h);
  k2_conv<<<dim3(5, BSZ), 256, 0, stream>>>(h, wt, bsum, yc);
  k3_mlp<<<NTOT / 16, 256, 0, stream>>>(yc, wl1t, bl1, wl2t, out);
}
